// MemNet_41566693491232
// MI455X (gfx1250) — matrix operand order repaired and checked by reading, not compiled, not run on hardware
//
#include <hip/hip_runtime.h>
#include <stdint.h>

#define BB 8
#define TT 256
#define EE 512
#define FF 2048
#define DD 256
#define NNM 2048      // memory slots N
#define VV 32000
#define LLN 2
#define HM_ 4
#define HD_ 64
#define HA_ 8
#define DH_ 64
#define IFX 1028      // 4*D + HM
#define MROWS (BB*TT) // 2048

typedef __attribute__((ext_vector_type(16))) __bf16 v16bf;
typedef __attribute__((ext_vector_type(8)))  float  v8f;
typedef __attribute__((ext_vector_type(4)))  unsigned int ui4;
typedef __attribute__((ext_vector_type(8)))  int i8v;
typedef __attribute__((ext_vector_type(4)))  int i4v;

// --- TDM availability (device pass only; host pass uses fallback path) ---
#if defined(__HIP_DEVICE_COMPILE__)
  #if __has_builtin(__builtin_amdgcn_tensor_load_to_lds) && __has_builtin(__builtin_amdgcn_s_wait_tensorcnt)
    #define USE_TDM 1
  #endif
#endif
#ifndef USE_TDM
  #define USE_TDM 0
#endif

#define PANEL_N   128          // B panel width (cols of C) staged per block
#define ROWSTRIDE 36           // ushorts per panel row in LDS: 16 DW data + 2 DW pad (18 banks)

__device__ __forceinline__ unsigned short f2bf(float f) {
  unsigned int u = __float_as_uint(f);
  u += 0x7FFFu + ((u >> 16) & 1u);        // round-to-nearest-even
  return (unsigned short)(u >> 16);
}
__device__ __forceinline__ float geluf(float x) {
  float x3 = x * x * x;
  return 0.5f * x * (1.f + tanhf(0.7978845608028654f * (x + 0.044715f * x3)));
}
__device__ __forceinline__ float sigmoidf_(float x) { return 1.f / (1.f + expf(-x)); }
__device__ __forceinline__ float softplus_clip(float x) {
  float sp = (x > 20.f) ? x : log1pf(expf(x));
  return fminf(fmaxf(sp, 1.f), 20.f);
}

#if USE_TDM
__device__ __forceinline__ void tdm_issue(ui4 g0, i8v g1) {
  i4v z = {};
#if __clang_major__ >= 23
  i8v z8 = {};
  __builtin_amdgcn_tensor_load_to_lds(g0, g1, z, z, z8, 0);
#else
  __builtin_amdgcn_tensor_load_to_lds(g0, g1, z, z, 0);
#endif
}
#endif

// Stage a B panel: rows x 32 K-slice of Bt (N x K bf16) into LDS with padded
// row stride (ROWSTRIDE). Rows beyond `rows` read as zero (TDM tensor_dim1 clip).
__device__ __forceinline__ void stage_b_panel(unsigned short* ldsDst,
                                              const unsigned short* gsrc,
                                              int rows, int K, int tid) {
#if USE_TDM
  if (tid < 32) {                 // wave 0 issues the tensor DMA (EXEC ignored by TDM)
    unsigned long long ga = (unsigned long long)(uintptr_t)gsrc;
    unsigned int lds = (unsigned int)(uintptr_t)ldsDst;
    ui4 g0;
    g0[0] = 1u;                                       // count=1, user descriptor
    g0[1] = lds;                                      // lds_addr
    g0[2] = (unsigned int)ga;                         // global_addr[31:0]
    g0[3] = (unsigned int)(ga >> 32) | (2u << 30);    // global_addr[56:32] | type=2
    i8v g1;
    // data_size=2B(1), pad_enable, pad_interval=16DW(3), pad_amount=2DW(1)
    g1[0] = (int)((1u << 16) | (1u << 20) | (3u << 22) | (1u << 25));
    g1[1] = (int)(32u << 16);                         // tensor_dim0 lo16 = 32
    g1[2] = (int)(((unsigned)rows & 0xFFFFu) << 16);  // td0 hi16=0 | tensor_dim1 lo16
    g1[3] = (int)(32u << 16);                         // td1 hi16=0 | tile_dim0=32
    g1[4] = (int)(unsigned)PANEL_N;                   // tile_dim1=128 | tile_dim2=0
    g1[5] = (int)(unsigned)K;                         // tensor_dim0_stride lo32
    g1[6] = 0;                                        // stride hi | dim1_stride lo
    g1[7] = 0;
    tdm_issue(g0, g1);
  }
#else
  // cooperative fallback: 128 rows x 16 dwords
  for (int u = tid; u < PANEL_N * 16; u += 256) {
    int row = u >> 4, kc = (u & 15) * 2;
    unsigned int v = 0;
    if (row < rows) v = *(const unsigned int*)(gsrc + (size_t)row * K + kc);
    *(unsigned int*)(ldsDst + row * ROWSTRIDE + kc) = v;
  }
#endif
}

// TENSORcnt is per-wave; only wave 0 issues TDM ops, so an unconditional wait
// is a zero-cost pass for waves 1..7 and avoids exec-mask save/restore in the
// hot loop.
__device__ __forceinline__ void tdm_wait_all(void) {
#if USE_TDM
  __builtin_amdgcn_s_wait_tensorcnt((short)0);
#endif
}

// ---------------------------------------------------------------------------
// WMMA GEMM: C[M,N] = act(A[M,K] @ B[K,N] + bias) + addend
// A bf16 row-major; Bt = B transposed (N x K) bf16.
// Block: 256 thr = 8 waves; block tile 64(M) x 128(N); wave tile 16(M) x 64(N)
// (4 accumulators, A fragment reused 4x). B panel staged in LDS via TDM
// (double buffered, s_wait_tensorcnt) with zero-padded ragged-N rows.
// ---------------------------------------------------------------------------
__global__ void gemm_wmma_bf16(const unsigned short* __restrict__ A,
                               const unsigned short* __restrict__ Bt,
                               const float* __restrict__ bias,
                               const float* __restrict__ addend,
                               float* __restrict__ Cf,
                               unsigned short* __restrict__ Cb,
                               int M, int N, int K, int act) {
  __shared__ unsigned short ldsB[2][PANEL_N * ROWSTRIDE];
  const int tid  = threadIdx.x;
  const int lane = tid & 31;
  const int wave = tid >> 5;
  const int wm = wave >> 1;              // 0..3  (M strip)
  const int wn = wave & 1;               // 0..1  (N half of panel)
  const int blockN = blockIdx.x * PANEL_N;
  const int mTile  = blockIdx.y * 64 + wm * 16;
  const int r  = lane & 15;
  const int kh = lane >> 4;
  const int rows = min(PANEL_N, N - blockN);

  union Frag { v16bf v; unsigned int u[8]; };
  v8f acc[4] = {};
  const unsigned short* ap = A + (size_t)(mTile + r) * K;

  stage_b_panel(&ldsB[0][0], Bt + (size_t)blockN * K, rows, K, tid);
  int cur = 0;
  for (int k0 = 0; k0 < K; k0 += 32) {
    tdm_wait_all();
    __syncthreads();                               // ldsB[cur] ready
    if (k0 + 32 < K)
      stage_b_panel(&ldsB[cur ^ 1][0], Bt + (size_t)blockN * K + (k0 + 32),
                    rows, K, tid);
    // A fragment (16-bit A 16x32 layout): lanes 0-15 K 0..7 & 16..23; 16-31 K 8..15 & 24..31
    Frag a;
    #pragma unroll
    for (int j = 0; j < 4; ++j)
      a.u[j] = *(const unsigned int*)(ap + k0 + 2 * j + 8 * kh);
    #pragma unroll
    for (int j = 4; j < 8; ++j)
      a.u[j] = *(const unsigned int*)(ap + k0 + 2 * (j - 4) + 8 * kh + 16);
    if (k0 + 32 < K) __builtin_prefetch(ap + k0 + 32, 0, 1);
    const unsigned short* lb = &ldsB[cur][0];
    #pragma unroll
    for (int s = 0; s < 4; ++s) {
      int pr = wn * 64 + s * 16 + r;               // panel row = C column offset
      const unsigned short* lbp = lb + pr * ROWSTRIDE + 8 * kh;
      Frag b;
      #pragma unroll
      for (int j = 0; j < 8; ++j)
        b.u[j] = *(const unsigned int*)(lbp + 2 * j + (j >= 4 ? 8 : 0));
      acc[s] = __builtin_amdgcn_wmma_f32_16x16x32_bf16(false, a.v, false, b.v,
                                                       (short)0, acc[s], false, false);
    }
    __syncthreads();                               // all reads of ldsB[cur] done
    cur ^= 1;
  }
  #pragma unroll
  for (int s = 0; s < 4; ++s) {
    int col = blockN + wn * 64 + s * 16 + r;
    if (col >= N) continue;
    #pragma unroll
    for (int j = 0; j < 8; ++j) {
      int row = mTile + j + 8 * kh;                // C/D: VGPR j -> M = j + 8*(lane>=16)
      float v = acc[s][j];
      if (bias)   v += bias[col];
      if (act)    v = geluf(v);
      if (addend) v += addend[(size_t)row * N + col];
      if (Cf) Cf[(size_t)row * N + col] = v;
      if (Cb) Cb[(size_t)row * N + col] = f2bf(v);
    }
  }
}

// f32 (K x N) -> bf16 transposed (N x K)
__global__ void k_convT(const float* __restrict__ W, unsigned short* __restrict__ Wt,
                        int K, int N) {
  size_t idx = (size_t)blockIdx.x * 256 + threadIdx.x;
  if (idx >= (size_t)N * K) return;
  int n = (int)(idx / K), k = (int)(idx % K);
  Wt[idx] = f2bf(W[(size_t)k * N + n]);
}

__global__ void k_embed(const int* __restrict__ ids, const float* __restrict__ emb,
                        const float* __restrict__ pos, float* __restrict__ x) {
  int idx = blockIdx.x * 256 + threadIdx.x;
  if (idx >= MROWS * EE) return;
  int row = idx / EE, e = idx % EE;
  int t = row % TT;
  x[idx] = emb[(size_t)ids[row] * EE + e] + pos[(size_t)t * EE + e];
}

// one block per row; f32 in, bf16 out
__global__ void k_layernorm(const float* __restrict__ x, const float* __restrict__ g,
                            const float* __restrict__ bta, unsigned short* __restrict__ h) {
  __shared__ float red[256];
  int row = blockIdx.x, tid = threadIdx.x;
  const float* xr = x + (size_t)row * EE;
  float x0 = xr[tid], x1 = xr[tid + 256];
  red[tid] = x0 + x1; __syncthreads();
  for (int s = 128; s > 0; s >>= 1) { if (tid < s) red[tid] += red[tid + s]; __syncthreads(); }
  float mean = red[0] * (1.f / EE); __syncthreads();
  float d0 = x0 - mean, d1 = x1 - mean;
  red[tid] = d0 * d0 + d1 * d1; __syncthreads();
  for (int s = 128; s > 0; s >>= 1) { if (tid < s) red[tid] += red[tid + s]; __syncthreads(); }
  float rs = rsqrtf(red[0] * (1.f / EE) + 1e-5f);
  h[(size_t)row * EE + tid]       = f2bf(d0 * rs * g[tid] + bta[tid]);
  h[(size_t)row * EE + tid + 256] = f2bf(d1 * rs * g[tid + 256] + bta[tid + 256]);
}

// one block per (b, head, query-pos); causal softmax attention
__global__ void k_attention(const float* __restrict__ q, const float* __restrict__ kk,
                            const float* __restrict__ vv, unsigned short* __restrict__ obf) {
  __shared__ float qv[DH_];
  __shared__ float red[256];
  __shared__ float aw[TT];
  int tid = threadIdx.x;
  int idx = blockIdx.x;                 // ((b*HA + h)*TT + qpos)
  int qpos = idx % TT;
  int bh = idx / TT;
  int h = bh % HA_, b = bh / HA_;
  const size_t base = (size_t)b * TT * EE + (size_t)h * DH_;
  if (tid < DH_) qv[tid] = q[base + (size_t)qpos * EE + tid];
  __syncthreads();
  int k = tid;
  float s = -1e30f;
  if (k <= qpos) {
    const float* kr = kk + base + (size_t)k * EE;
    float d = 0.f;
    #pragma unroll 8
    for (int e = 0; e < DH_; ++e) d += qv[e] * kr[e];
    s = d * 0.125f;                      // 1/sqrt(64)
  }
  red[tid] = s; __syncthreads();
  for (int st = 128; st > 0; st >>= 1) { if (tid < st) red[tid] = fmaxf(red[tid], red[tid + st]); __syncthreads(); }
  float mx = red[0]; __syncthreads();
  float e = (k <= qpos) ? expf(s - mx) : 0.f;
  red[tid] = e; __syncthreads();
  for (int st = 128; st > 0; st >>= 1) { if (tid < st) red[tid] += red[tid + st]; __syncthreads(); }
  float inv = 1.f / red[0];
  aw[tid] = e * inv;
  __syncthreads();
  if (tid < DH_) {
    float o = 0.f;
    for (int kx = 0; kx <= qpos; ++kx)
      o += aw[kx] * vv[base + (size_t)kx * EE + tid];
    obf[base + (size_t)qpos * EE + tid] = f2bf(o);
  }
}

// i = iface[:,t,:] + rv @ W_if_r + b_if   (B x IF)
__global__ void k_iface_step(const float* __restrict__ iface, const float* __restrict__ RV,
                             const float* __restrict__ Wifr, const float* __restrict__ b_if,
                             float* __restrict__ ivec, int t) {
  int idx = blockIdx.x * 256 + threadIdx.x;
  if (idx >= BB * IFX) return;
  int b = idx / IFX, j = idx % IFX;
  const float* rv = RV + ((size_t)b * TT + t) * DD;
  float acc = b_if[j] + iface[((size_t)b * TT + t) * IFX + j];
  for (int d = 0; d < DD; ++d) acc += rv[d] * Wifr[(size_t)d * IFX + j];
  ivec[idx] = acc;
}

// one block per (b, head): cosine sims over all N slots, top-8 threshold softmax
// for read & write heads, rv_new gather, sparse erase/write mem update.
__global__ void k_mem_step(float* __restrict__ mem, const float* __restrict__ ivec,
                           float* __restrict__ RV, const float* __restrict__ betaRp,
                           const float* __restrict__ betaWp, int t) {
  __shared__ float simr[NNM];
  __shared__ float simw[NNM];
  __shared__ float kr[HD_], kw[HD_];
  __shared__ float sval[256];
  __shared__ int   sidx[256];
  __shared__ int   lidx[16];       // [0..7] read top-8, [8..15] write top-8
  __shared__ float lval[16];
  __shared__ float lw[16];
  int tid = threadIdx.x;
  int h = blockIdx.x % HM_, b = blockIdx.x / HM_;
  const float* iv = ivec + (size_t)b * IFX;
  if (tid < HD_) { kr[tid] = iv[h * HD_ + tid]; kw[tid] = iv[DD + h * HD_ + tid]; }
  __syncthreads();
  if (tid == 0) {
    float sr = 0.f, sw = 0.f;
    for (int d = 0; d < HD_; ++d) { sr += kr[d] * kr[d]; sw += kw[d] * kw[d]; }
    sval[0] = rsqrtf(sr + 1e-8f); sval[1] = rsqrtf(sw + 1e-8f);
  }
  __syncthreads();
  float nr = sval[0], nw = sval[1];
  __syncthreads();
  if (tid < HD_) { kr[tid] *= nr; kw[tid] *= nw; }
  __syncthreads();
  const size_t mbase = (size_t)b * NNM * DD + (size_t)h * HD_;  // mem[b,n,h,d]
  for (int n = tid; n < NNM; n += 256) {
    const float* mr = mem + mbase + (size_t)n * DD;
    float ssq = 0.f, dr = 0.f, dw = 0.f;
    #pragma unroll 8
    for (int d = 0; d < HD_; ++d) { float m = mr[d]; ssq += m * m; dr += m * kr[d]; dw += m * kw[d]; }
    float inv = rsqrtf(ssq + 1e-8f);
    simr[n] = dr * inv; simw[n] = dw * inv;
  }
  __syncthreads();
  for (int pass = 0; pass < 2; ++pass) {
    float* sim = pass ? simw : simr;
    for (int sel = 0; sel < 8; ++sel) {
      float mv = -3e38f; int mi = 0;
      for (int n = tid; n < NNM; n += 256) { float v = sim[n]; if (v > mv) { mv = v; mi = n; } }
      sval[tid] = mv; sidx[tid] = mi; __syncthreads();
      for (int s = 128; s > 0; s >>= 1) {
        if (tid < s && sval[tid + s] > sval[tid]) { sval[tid] = sval[tid + s]; sidx[tid] = sidx[tid + s]; }
        __syncthreads();
      }
      if (tid == 0) {
        int slot = pass * 8 + sel;
        lidx[slot] = sidx[0]; lval[slot] = sval[0];
        sim[sidx[0]] = -3e38f;
      }
      __syncthreads();
    }
  }
  if (tid < 2) {  // softmax over selected (thread 0: read head, 1: write head)
    float beta = softplus_clip(tid == 0 ? betaRp[0] : betaWp[0]);
    int o = tid * 8;
    float m = lval[o];                 // first selected is global max
    float s = 0.f;
    for (int i = 0; i < 8; ++i) { float e = expf(beta * (lval[o + i] - m)); lw[o + i] = e; s += e; }
    float inv = 1.f / s;
    for (int i = 0; i < 8; ++i) lw[o + i] *= inv;
  }
  __syncthreads();
  if (tid < HD_) {
    int d = tid;
    float rv = 0.f;
    for (int i = 0; i < 8; ++i) rv += lw[i] * mem[mbase + (size_t)lidx[i] * DD + d];
    if (t + 1 < TT) RV[((size_t)b * TT + (t + 1)) * DD + h * HD_ + d] = rv;
    float er = sigmoidf_(iv[3 * DD + h * HD_ + d]);
    float wv = iv[2 * DD + h * HD_ + d];
    float gv = sigmoidf_(iv[4 * DD + h]) * wv;
    for (int i = 8; i < 16; ++i) {
      size_t off = mbase + (size_t)lidx[i] * DD + d;
      float w = lw[i];
      float m = mem[off];
      mem[off] = m * (1.f - w * er) + w * gv;
    }
  }
}

__global__ void k_mem_init(float* __restrict__ mem, const float* __restrict__ mem_init) {
  size_t idx = (size_t)blockIdx.x * 256 + threadIdx.x;
  if (idx >= (size_t)BB * NNM * DD) return;
  mem[idx] = mem_init[idx % ((size_t)NNM * DD)];
}
__global__ void k_rv0(float* __restrict__ RV) {
  int idx = blockIdx.x * 256 + threadIdx.x;
  if (idx >= BB * DD) return;
  RV[((size_t)(idx / DD)) * TT * DD + (idx % DD)] = 0.f;
}
__global__ void k_f32_to_bf16(const float* __restrict__ x, unsigned short* __restrict__ y, size_t n) {
  size_t idx = (size_t)blockIdx.x * 256 + threadIdx.x;
  if (idx < n) y[idx] = f2bf(x[idx]);
}

// ---------------------------------------------------------------------------
extern "C" void kernel_launch(void* const* d_in, const int* in_sizes, int n_in,
                              void* d_out, int out_size, void* d_ws, size_t ws_size,
                              hipStream_t stream) {
  const int*   ids    = (const int*)d_in[0];
  const float* emb    = (const float*)d_in[1];
  const float* pos    = (const float*)d_in[2];
  const float* ln1_g  = (const float*)d_in[3];
  const float* ln1_b  = (const float*)d_in[4];
  const float* ln2_g  = (const float*)d_in[5];
  const float* ln2_b  = (const float*)d_in[6];
  const float* Wq     = (const float*)d_in[7];
  const float* Wk     = (const float*)d_in[8];
  const float* Wv     = (const float*)d_in[9];
  const float* Wo     = (const float*)d_in[10];
  const float* W1     = (const float*)d_in[11];
  const float* b1     = (const float*)d_in[12];
  const float* W2     = (const float*)d_in[13];
  const float* b2     = (const float*)d_in[14];
  const float* lnf_g  = (const float*)d_in[15];
  const float* lnf_b  = (const float*)d_in[16];
  const float* W_if_h = (const float*)d_in[17];
  const float* W_if_r = (const float*)d_in[18];
  const float* b_if   = (const float*)d_in[19];
  const float* W_lg_h = (const float*)d_in[20];
  const float* W_lg_r = (const float*)d_in[21];
  const float* b_lg   = (const float*)d_in[22];
  const float* betaR  = (const float*)d_in[23];
  const float* betaW  = (const float*)d_in[24];
  const float* memini = (const float*)d_in[25];
  float* out = (float*)d_out;

  char* ws = (char*)d_ws;
  size_t off = 0;
  auto alloc = [&](size_t bytes) -> void* {
    void* p = ws + off;
    off += (bytes + 255) & ~(size_t)255;
    return p;
  };
  // bf16 transposed weights
  unsigned short* WqT   = (unsigned short*)alloc((size_t)LLN * EE * EE * 2);
  unsigned short* WkT   = (unsigned short*)alloc((size_t)LLN * EE * EE * 2);
  unsigned short* WvT   = (unsigned short*)alloc((size_t)LLN * EE * EE * 2);
  unsigned short* WoT   = (unsigned short*)alloc((size_t)LLN * EE * EE * 2);
  unsigned short* W1T   = (unsigned short*)alloc((size_t)LLN * FF * EE * 2);
  unsigned short* W2T   = (unsigned short*)alloc((size_t)LLN * EE * FF * 2);
  unsigned short* WifhT = (unsigned short*)alloc((size_t)IFX * EE * 2);
  unsigned short* WlghT = (unsigned short*)alloc((size_t)VV * EE * 2);
  unsigned short* WlgrT = (unsigned short*)alloc((size_t)VV * DD * 2);
  // activations
  float*          xbuf  = (float*)alloc((size_t)MROWS * EE * 4);
  unsigned short* hbuf  = (unsigned short*)alloc((size_t)MROWS * EE * 2);
  float*          qf    = (float*)alloc((size_t)MROWS * EE * 4);
  float*          kf    = (float*)alloc((size_t)MROWS * EE * 4);
  float*          vf    = (float*)alloc((size_t)MROWS * EE * 4);
  unsigned short* obf   = (unsigned short*)alloc((size_t)MROWS * EE * 2);
  unsigned short* midbf = (unsigned short*)alloc((size_t)MROWS * FF * 2);
  float*          ifc   = (float*)alloc((size_t)MROWS * IFX * 4);
  float*          ivec  = (float*)alloc((size_t)BB * IFX * 4);
  float*          memb  = (float*)alloc((size_t)BB * NNM * DD * 4);
  float*          RV    = (float*)alloc((size_t)MROWS * DD * 4);
  unsigned short* rvbf  = (unsigned short*)alloc((size_t)MROWS * DD * 2);

  auto cgrid = [](size_t n) { return (unsigned)((n + 255) / 256); };

  // --- weight conversion (f32 KxN -> bf16 NxK) ---
  for (int l = 0; l < LLN; ++l) {
    k_convT<<<cgrid((size_t)EE * EE), 256, 0, stream>>>(Wq + (size_t)l * EE * EE, WqT + (size_t)l * EE * EE, EE, EE);
    k_convT<<<cgrid((size_t)EE * EE), 256, 0, stream>>>(Wk + (size_t)l * EE * EE, WkT + (size_t)l * EE * EE, EE, EE);
    k_convT<<<cgrid((size_t)EE * EE), 256, 0, stream>>>(Wv + (size_t)l * EE * EE, WvT + (size_t)l * EE * EE, EE, EE);
    k_convT<<<cgrid((size_t)EE * EE), 256, 0, stream>>>(Wo + (size_t)l * EE * EE, WoT + (size_t)l * EE * EE, EE, EE);
    k_convT<<<cgrid((size_t)EE * FF), 256, 0, stream>>>(W1 + (size_t)l * EE * FF, W1T + (size_t)l * FF * EE, EE, FF);
    k_convT<<<cgrid((size_t)FF * EE), 256, 0, stream>>>(W2 + (size_t)l * FF * EE, W2T + (size_t)l * EE * FF, FF, EE);
  }
  k_convT<<<cgrid((size_t)EE * IFX), 256, 0, stream>>>(W_if_h, WifhT, EE, IFX);
  k_convT<<<cgrid((size_t)EE * VV), 256, 0, stream>>>(W_lg_h, WlghT, EE, VV);
  k_convT<<<cgrid((size_t)DD * VV), 256, 0, stream>>>(W_lg_r, WlgrT, DD, VV);

  // --- embedding ---
  k_embed<<<cgrid((size_t)MROWS * EE), 256, 0, stream>>>(ids, emb, pos, xbuf);

  auto ggrid = [](int M, int N) { return dim3((N + PANEL_N - 1) / PANEL_N, (M + 63) / 64); };

  // --- transformer layers ---
  for (int l = 0; l < LLN; ++l) {
    k_layernorm<<<MROWS, 256, 0, stream>>>(xbuf, ln1_g + (size_t)l * EE, ln1_b + (size_t)l * EE, hbuf);
    gemm_wmma_bf16<<<ggrid(MROWS, EE), 256, 0, stream>>>(hbuf, WqT + (size_t)l * EE * EE, nullptr, nullptr, qf, nullptr, MROWS, EE, EE, 0);
    gemm_wmma_bf16<<<ggrid(MROWS, EE), 256, 0, stream>>>(hbuf, WkT + (size_t)l * EE * EE, nullptr, nullptr, kf, nullptr, MROWS, EE, EE, 0);
    gemm_wmma_bf16<<<ggrid(MROWS, EE), 256, 0, stream>>>(hbuf, WvT + (size_t)l * EE * EE, nullptr, nullptr, vf, nullptr, MROWS, EE, EE, 0);
    k_attention<<<BB * HA_ * TT, 256, 0, stream>>>(qf, kf, vf, obf);
    gemm_wmma_bf16<<<ggrid(MROWS, EE), 256, 0, stream>>>(obf, WoT + (size_t)l * EE * EE, nullptr, xbuf, xbuf, nullptr, MROWS, EE, EE, 0);
    k_layernorm<<<MROWS, 256, 0, stream>>>(xbuf, ln2_g + (size_t)l * EE, ln2_b + (size_t)l * EE, hbuf);
    gemm_wmma_bf16<<<ggrid(MROWS, FF), 256, 0, stream>>>(hbuf, W1T + (size_t)l * FF * EE, b1 + (size_t)l * FF, nullptr, nullptr, midbf, MROWS, FF, EE, 1);
    gemm_wmma_bf16<<<ggrid(MROWS, EE), 256, 0, stream>>>(midbf, W2T + (size_t)l * EE * FF, b2 + (size_t)l * EE, xbuf, xbuf, nullptr, MROWS, EE, FF, 0);
  }

  // --- final LN + interface + vocab logits ---
  k_layernorm<<<MROWS, 256, 0, stream>>>(xbuf, lnf_g, lnf_b, hbuf);
  gemm_wmma_bf16<<<ggrid(MROWS, IFX), 256, 0, stream>>>(hbuf, WifhT, nullptr, nullptr, ifc, nullptr, MROWS, IFX, EE, 0);
  gemm_wmma_bf16<<<ggrid(MROWS, VV), 256, 0, stream>>>(hbuf, WlghT, b_lg, nullptr, out, nullptr, MROWS, VV, EE, 0);

  // --- memory scan (rv carries recorded; big log_r GEMM hoisted to the end) ---
  k_mem_init<<<cgrid((size_t)BB * NNM * DD), 256, 0, stream>>>(memb, memini);
  k_rv0<<<cgrid((size_t)BB * DD), 256, 0, stream>>>(RV);
  for (int t = 0; t < TT; ++t) {
    k_iface_step<<<cgrid((size_t)BB * IFX), 256, 0, stream>>>(ifc, RV, W_if_r, b_if, ivec, t);
    k_mem_step<<<BB * HM_, 256, 0, stream>>>(memb, ivec, RV, betaR, betaW, t);
  }
  k_f32_to_bf16<<<cgrid((size_t)MROWS * DD), 256, 0, stream>>>(RV, rvbf, (size_t)MROWS * DD);
  gemm_wmma_bf16<<<ggrid(MROWS, VV), 256, 0, stream>>>(rvbf, WlgrT, nullptr, out, out, nullptr, MROWS, VV, DD, 0);
}
